// EncodeBlock_16509854286199
// MI455X (gfx1250) — compile-verified, matrix operand order checked by reading, not run on hardware
//
#include <hip/hip_runtime.h>
#include <math.h>

// ---------------------------------------------------------------------------
// Problem constants
// ---------------------------------------------------------------------------
#define BWIN 1024
#define NTOK 64
#define CDIM 256
#define HEADS 8
#define DHEAD 32
#define NANC 16
#define LOG100 4.605170185988092f

typedef __attribute__((ext_vector_type(16))) _Float16 v16h;
typedef __attribute__((ext_vector_type(8)))  float    v8f;

// ---------------------------------------------------------------------------
// LDS layout (bytes). Total 296KB dynamic LDS (<= 320KB per workgroup).
// ---------------------------------------------------------------------------
#define OFF_QN    0        // half [8][64][32]  (later reused as sOut)
#define OFF_KN    32768    // half [8][64][32]
#define OFF_SVN   65536    // half [8][64][32]
#define OFF_AN    98304    // half [8][16][32]
#define OFF_A1N   106496   // half [8][64][32]  (phase1: sX lives here)
#define OFF_A2N   139264   // half [8][64][32]  (phase1: sL + wave scratch)
#define OFF_ATTN  172032   // float [8][64][64]
#define LDS_BYTES 303104
// phase-1 temporaries aliased into the A1N/A2N region:
#define OFF_SX    106496   // half [64][256]  32KB
#define OFF_SL    139264   // half [16][256]  8KB
#define OFF_SCR   147456   // float [8][16*32] 16KB
// async f32 staging buffers, aliased into the ATTN region (free in phase 1):
#define OFF_XF32  OFF_ATTN            // float [64][256]  64KB
#define OFF_LF32  (OFF_ATTN + 65536)  // float [16][256]  16KB

// ---------------------------------------------------------------------------
// CDNA5 async DMA helpers (ASYNCcnt-tracked global->LDS copies)
// ---------------------------------------------------------------------------
__device__ __forceinline__ void async_copy16(unsigned lds_byte_off,
                                             const void* gaddr) {
  // GLOBAL_LOAD_ASYNC_TO_LDS_B128: VDST = LDS byte address, VADDR = 64-bit VA
  asm volatile("global_load_async_to_lds_b128 %0, %1, off"
               :
               : "v"(lds_byte_off), "v"(gaddr)
               : "memory");
}
__device__ __forceinline__ void wait_async0() {
  asm volatile("s_wait_asynccnt 0x0" ::: "memory");
}

// ---------------------------------------------------------------------------
// WMMA helpers (CDNA5 wave32, V_WMMA_F32_16X16X32_F16)
// ---------------------------------------------------------------------------
__device__ __forceinline__ v8f wmma16(v16h a, v16h b, v8f c) {
  return __builtin_amdgcn_wmma_f32_16x16x32_f16(false, a, false, b, (short)0, c,
                                                false, false);
}

// A fragment (16x32 MxK) from a row-major f16 matrix with leading dim ld.
// Per ISA layout: lane<16 -> K {k0..k0+7, k0+16..k0+23}, lane>=16 -> +8.
// Also used for B fragments when B^T is stored row-major (col <- row0).
__device__ __forceinline__ v16h load_frag_f16(const _Float16* S, int row0,
                                              int k0, int ld) {
  const int lane = threadIdx.x & 31;
  const _Float16* p = S + (size_t)(row0 + (lane & 15)) * ld + k0 +
                      ((lane & 16) ? 8 : 0);
  v16h r;
#pragma unroll
  for (int e = 0; e < 8; e++) { r[e] = p[e]; r[e + 8] = p[16 + e]; }
  return r;
}

// Same fragment but source is f32 (converted per element).
__device__ __forceinline__ v16h load_frag_f32(const float* S, int row0, int k0,
                                              int ld) {
  const int lane = threadIdx.x & 31;
  const float* p = S + (size_t)(row0 + (lane & 15)) * ld + k0 +
                   ((lane & 16) ? 8 : 0);
  v16h r;
#pragma unroll
  for (int e = 0; e < 8; e++) {
    r[e] = (_Float16)p[e];
    r[e + 8] = (_Float16)p[16 + e];
  }
  return r;
}

// B fragment from a row-major [K][N] f16 matrix (element B(k,n)).
__device__ __forceinline__ v16h load_frag_brm(const _Float16* Bm, int col0,
                                              int k0, int ld) {
  const int lane = threadIdx.x & 31;
  const int n = col0 + (lane & 15);
  const int kb = k0 + ((lane & 16) ? 8 : 0);
  v16h r;
#pragma unroll
  for (int e = 0; e < 8; e++) {
    r[e] = Bm[(size_t)(kb + e) * ld + n];
    r[e + 8] = Bm[(size_t)(kb + 16 + e) * ld + n];
  }
  return r;
}

// Store a 16x16 f32 accumulator tile: M = vgpr + 8*(lane>=16), N = lane&15.
__device__ __forceinline__ void store_frag(float* S, v8f d, int row0, int col0,
                                           int ld) {
  const int lane = threadIdx.x & 31;
  const int col = col0 + (lane & 15);
  const int rb = row0 + ((lane & 16) ? 8 : 0);
#pragma unroll
  for (int v = 0; v < 8; v++) S[(size_t)(rb + v) * ld + col] = d[v];
}

// ---------------------------------------------------------------------------
// Prep kernels
// ---------------------------------------------------------------------------
__global__ void k_f32_to_f16(const float* __restrict__ src,
                             _Float16* __restrict__ dst, int n) {
  int i = blockIdx.x * 256 + threadIdx.x;
  if (i < n) dst[i] = (_Float16)src[i];
}

// CPB MLP: tab[t][h] for t in [0,225), h in [0,8)
__global__ void k_cpb(const float* __restrict__ w1, const float* __restrict__ b1,
                      const float* __restrict__ w2, float* __restrict__ tab) {
  int idx = blockIdx.x * 256 + threadIdx.x;
  if (idx >= 225 * HEADS) return;
  int t = idx >> 3, h = idx & 7;
  float r0 = (float)(t / 15 - 7);
  float r1 = (float)(t % 15 - 7);
  float g0 = r0 / 7.0f * 8.0f;
  float g1 = r1 / 7.0f * 8.0f;
  float c0 = copysignf(log2f(fabsf(g0) + 1.0f) / 3.0f, g0);
  float c1 = copysignf(log2f(fabsf(g1) + 1.0f) / 3.0f, g1);
  float acc = 0.0f;
  for (int j = 0; j < 512; j++) {
    float hd = c0 * w1[j * 2] + c1 * w1[j * 2 + 1] + b1[j];
    hd = fmaxf(hd, 0.0f);
    acc += hd * w2[h * 512 + j];
  }
  tab[t * 8 + h] = acc;
}

// per-head attention scales: m -> exp(min(ls,log100)); p -> (exp(...))^2
__global__ void k_scales(const float* __restrict__ m_ls,
                         const float* __restrict__ p_ls,
                         float* __restrict__ scales) {
  int i = threadIdx.x;
  if (i < HEADS) {
    scales[i] = __expf(fminf(m_ls[i], LOG100));
  } else if (i < 2 * HEADS) {
    float e = __expf(fminf(p_ls[i - HEADS], LOG100));
    scales[i] = e * e;
  }
}

// ---------------------------------------------------------------------------
// Fused per-window attention kernel. grid.x = BWIN, block = 256 (8 waves).
// ---------------------------------------------------------------------------
__global__ void attn_fused(const float* __restrict__ x, const float* __restrict__ l,
                           const float* __restrict__ s,
                           const float* __restrict__ q_bias,
                           const float* __restrict__ v_bias,
                           const _Float16* __restrict__ qkv_w,
                           const _Float16* __restrict__ anchor_w,
                           const _Float16* __restrict__ proj_w,
                           const float* __restrict__ proj_b,
                           const float* __restrict__ tab,
                           const float* __restrict__ scales,
                           float* __restrict__ out) {
  extern __shared__ char lds[];
  const int b = blockIdx.x;
  const int tid = threadIdx.x;
  const int wave = tid >> 5;
  const int lane = tid & 31;

  _Float16* sX = (_Float16*)(lds + OFF_SX);  // [64][256]
  _Float16* sL = (_Float16*)(lds + OFF_SL);  // [16][256]

  // ---- stage inputs: async DMA (global -> LDS, ASYNCcnt), then f32 -> f16 ----
  const float* xg = x + (size_t)b * (NTOK * CDIM);
  const float* lg = l + (size_t)b * (NANC * CDIM);
#pragma unroll
  for (int i = 0; i < (NTOK * CDIM) / 4 / 256; i++) {  // 16 x 16B per thread
    int c = tid + i * 256;
    async_copy16((unsigned)(OFF_XF32 + c * 16), (const void*)(xg + c * 4));
  }
#pragma unroll
  for (int i = 0; i < (NANC * CDIM) / 4 / 256; i++) {  // 4 x 16B per thread
    int c = tid + i * 256;
    async_copy16((unsigned)(OFF_LF32 + c * 16), (const void*)(lg + c * 4));
  }
  wait_async0();
  __syncthreads();
  const float* xf = (const float*)(lds + OFF_XF32);
  const float* lf = (const float*)(lds + OFF_LF32);
  for (int i = tid; i < NTOK * CDIM; i += 256) sX[i] = (_Float16)xf[i];
  for (int i = tid; i < NANC * CDIM; i += 256) sL[i] = (_Float16)lf[i];
  __syncthreads();

  // ---- phase 1: QKV + anchor GEMM, bias, l2-normalize, v*s ----
  // 104 units of 16 rows x 32 cols (one head's D-slice):
  //   u in [0,32): q, [32,64): k, [64,96): v, [96,104): anchor
  float* scr = (float*)(lds + OFF_SCR) + wave * (16 * 32);
  for (int u = wave; u < 104; u += 8) {
    int type, h, rt;
    if (u < 96) { type = u >> 5; int r = u & 31; h = r >> 2; rt = r & 3; }
    else        { type = 3; h = u - 96; rt = 0; }
    const _Float16* src;
    const _Float16* w;
    int row0;
    if (type == 3) { src = sL; row0 = 0;       w = anchor_w + (size_t)(h * DHEAD) * CDIM; }
    else           { src = sX; row0 = rt * 16; w = qkv_w + (size_t)(type * CDIM + h * DHEAD) * CDIM; }

    // hint the weight slice (32 rows x 256 cols f16 = 16KB) toward the caches
    __builtin_prefetch(w, 0, 1);
    __builtin_prefetch(w + 16 * CDIM, 0, 1);

    v8f acc0 = {};
    v8f acc1 = {};
#pragma unroll
    for (int kk = 0; kk < CDIM; kk += 32) {
      v16h af = load_frag_f16(src, row0, kk, CDIM);
      v16h b0 = load_frag_f16(w, 0, kk, CDIM);    // W^T cols 0..15 of slice
      v16h b1 = load_frag_f16(w, 16, kk, CDIM);   // cols 16..31
      acc0 = wmma16(af, b0, acc0);
      acc1 = wmma16(af, b1, acc1);
    }
    store_frag(scr, acc0, 0, 0, 32);
    store_frag(scr, acc1, 0, 16, 32);

    if (lane < 16) {
      float vals[32];
      if (type == 2) {  // v: + v_bias, * s  -> svn (f16)
        int n = row0 + lane;
        const float* sg = s + ((size_t)b * NTOK + n) * CDIM + h * DHEAD;
        _Float16* dst = (_Float16*)(lds + OFF_SVN) + (size_t)(h * 64 + n) * 32;
#pragma unroll
        for (int d = 0; d < 32; d++) {
          float v = scr[lane * 32 + d] + v_bias[h * DHEAD + d];
          dst[d] = (_Float16)(v * sg[d]);
        }
      } else {          // q/k/anchor: (+q_bias for q), l2 normalize over D
        float ss = 0.0f;
#pragma unroll
        for (int d = 0; d < 32; d++) {
          float v = scr[lane * 32 + d];
          if (type == 0) v += q_bias[h * DHEAD + d];
          vals[d] = v;
          ss += v * v;
        }
        float inv = 1.0f / fmaxf(sqrtf(ss), 1e-12f);
        _Float16* dst;
        if (type == 0)      dst = (_Float16*)(lds + OFF_QN) + (size_t)(h * 64 + row0 + lane) * 32;
        else if (type == 1) dst = (_Float16*)(lds + OFF_KN) + (size_t)(h * 64 + row0 + lane) * 32;
        else                dst = (_Float16*)(lds + OFF_AN) + (size_t)(h * 16 + lane) * 32;
#pragma unroll
        for (int d = 0; d < 32; d++) dst[d] = (_Float16)(vals[d] * inv);
      }
    }
  }
  __syncthreads();

  // ---- phase 2: per-head attention (wave h owns head h) ----
  {
    const int h = wave;
    _Float16* qn_h  = (_Float16*)(lds + OFF_QN)  + (size_t)h * 64 * 32;
    _Float16* kn_h  = (_Float16*)(lds + OFF_KN)  + (size_t)h * 64 * 32;
    _Float16* svn_h = (_Float16*)(lds + OFF_SVN) + (size_t)h * 64 * 32;
    _Float16* an_h  = (_Float16*)(lds + OFF_AN)  + (size_t)h * 16 * 32;
    _Float16* a1n_h = (_Float16*)(lds + OFF_A1N) + (size_t)h * 64 * 32;
    _Float16* a2n_h = (_Float16*)(lds + OFF_A2N) + (size_t)h * 64 * 32;
    float*    attn_h = (float*)(lds + OFF_ATTN)  + (size_t)h * 64 * 64;

    // a1 = l2n(q)@an^T, a2 = l2n(k)@an^T ; row-normalize over NA=16, pad K->32
    for (int t = 0; t < 8; t++) {
      const _Float16* src = (t < 4) ? qn_h : kn_h;
      _Float16* dst = (t < 4) ? a1n_h : a2n_h;
      int row0 = (t & 3) * 16;
      v16h af = load_frag_f16(src, row0, 0, 32);
      v16h bf = load_frag_f16(an_h, 0, 0, 32);  // B(k,n) = an[n][k]
      v8f acc = {};
      acc = wmma16(af, bf, acc);
      store_frag(attn_h, acc, 0, 0, 16);        // 16x16 scratch
      if (lane < 16) {
        float vv[16];
        float ss = 0.0f;
#pragma unroll
        for (int m = 0; m < 16; m++) {
          float v = attn_h[lane * 16 + m];
          vv[m] = v;
          ss += v * v;
        }
        float inv = 1.0f / fmaxf(sqrtf(ss), 1e-12f);
        _Float16* dr = dst + (size_t)(row0 + lane) * 32;
#pragma unroll
        for (int m = 0; m < 16; m++) dr[m] = (_Float16)(vv[m] * inv);
#pragma unroll
        for (int m = 16; m < 32; m++) dr[m] = (_Float16)0.0f;
      }
    }

    // attn = a1n @ a2n^T, scaled, + relative position bias
    const float scale = scales[h];
    for (int t = 0; t < 16; t++) {
      int row0 = (t >> 2) * 16, col0 = (t & 3) * 16;
      v16h af = load_frag_f16(a1n_h, row0, 0, 32);
      v16h bf = load_frag_f16(a2n_h, col0, 0, 32);
      v8f acc = {};
      acc = wmma16(af, bf, acc);
      int col = col0 + (lane & 15);
      int rb = row0 + ((lane & 16) ? 8 : 0);
#pragma unroll
      for (int v = 0; v < 8; v++) {
        int row = rb + v;
        int di = (row >> 3) - (col >> 3) + 7;
        int dj = (row & 7) - (col & 7) + 7;
        float tb = tab[(di * 15 + dj) * 8 + h];
        float rpb = 16.0f / (1.0f + __expf(-tb));
        attn_h[row * 64 + col] = acc[v] * scale + rpb;
      }
    }

    // softmax in-place (f32), each lane owns rows lane and lane+32
#pragma unroll
    for (int rr = 0; rr < 2; rr++) {
      int row = lane + rr * 32;
      float* ar = attn_h + row * 64;
      float mx = -1e30f;
      for (int c = 0; c < 64; c++) mx = fmaxf(mx, ar[c]);
      float sum = 0.0f;
      for (int c = 0; c < 64; c++) {
        float e = __expf(ar[c] - mx);
        ar[c] = e;
        sum += e;
      }
      float inv = 1.0f / sum;
      for (int c = 0; c < 64; c++) ar[c] *= inv;
    }

    // out_h = probs(64x64, read as f32) @ svn_h(64x32); overwrite qn region
    _Float16* sOut_h = (_Float16*)(lds + OFF_QN) + (size_t)h * 64 * 32;
    for (int t = 0; t < 8; t++) {
      int row0 = (t >> 1) * 16, col0 = (t & 1) * 16;
      v8f acc = {};
#pragma unroll
      for (int kk = 0; kk < 64; kk += 32) {
        v16h af = load_frag_f32(attn_h, row0, kk, 64);
        v16h bf = load_frag_brm(svn_h, col0, kk, 32);
        acc = wmma16(af, bf, acc);
      }
      int col = col0 + (lane & 15);
      int rb = row0 + ((lane & 16) ? 8 : 0);
#pragma unroll
      for (int v = 0; v < 8; v++)
        sOut_h[(size_t)(rb + v) * 32 + col] = (_Float16)acc[v];
    }
  }
  __syncthreads();

  // ---- phase 3: projection: Y = out @ proj_w^T + proj_b ----
  // sOut is head-major [8][64][32]; logical column c = h*32 + d.
  const _Float16* sOut = (const _Float16*)(lds + OFF_QN);
  float* og = out + (size_t)b * (NTOK * CDIM);
  for (int t = wave; t < 64; t += 8) {
    int row0 = (t >> 4) * 16, col0 = (t & 15) * 16;
    v8f acc = {};
#pragma unroll
    for (int kk = 0; kk < CDIM; kk += 32) {
      // k-window [kk,kk+32) lies inside head kk/32 of sOut
      v16h af = load_frag_f16(sOut + (size_t)(kk >> 5) * 64 * 32, row0, 0, 32);
      v16h bf = load_frag_f16(proj_w, col0, kk, CDIM);
      acc = wmma16(af, bf, acc);
    }
    int col = col0 + (lane & 15);
    int rb = row0 + ((lane & 16) ? 8 : 0);
    float pb = proj_b[col];
#pragma unroll
    for (int v = 0; v < 8; v++)
      og[(size_t)(rb + v) * CDIM + col] = acc[v] + pb;
  }
}

// ---------------------------------------------------------------------------
// Host launcher
// ---------------------------------------------------------------------------
extern "C" void kernel_launch(void* const* d_in, const int* in_sizes, int n_in,
                              void* d_out, int out_size, void* d_ws,
                              size_t ws_size, hipStream_t stream) {
  (void)in_sizes; (void)n_in; (void)out_size; (void)ws_size;

  const float* m        = (const float*)d_in[0];
  const float* p        = (const float*)d_in[1];
  const float* m_l      = (const float*)d_in[2];
  const float* m_s      = (const float*)d_in[3];
  const float* p_l      = (const float*)d_in[4];
  const float* p_s      = (const float*)d_in[5];
  const float* m_qkv_w  = (const float*)d_in[6];
  const float* p_qkv_w  = (const float*)d_in[7];
  const float* m_q_bias = (const float*)d_in[8];
  const float* m_v_bias = (const float*)d_in[9];
  const float* p_q_bias = (const float*)d_in[10];
  const float* p_v_bias = (const float*)d_in[11];
  const float* m_ls     = (const float*)d_in[12];
  const float* p_ls     = (const float*)d_in[13];
  const float* cpb_w1   = (const float*)d_in[14];
  const float* cpb_b1   = (const float*)d_in[15];
  const float* cpb_w2   = (const float*)d_in[16];
  const float* m_anc_w  = (const float*)d_in[17];
  const float* p_anc_w  = (const float*)d_in[18];
  const float* m_proj_w = (const float*)d_in[19];
  const float* m_proj_b = (const float*)d_in[20];
  const float* p_proj_w = (const float*)d_in[21];
  const float* p_proj_b = (const float*)d_in[22];

  char* ws = (char*)d_ws;
  _Float16* w_mqkv = (_Float16*)(ws + 0);
  _Float16* w_pqkv = (_Float16*)(ws + 393216);
  _Float16* w_manc = (_Float16*)(ws + 786432);
  _Float16* w_panc = (_Float16*)(ws + 917504);
  _Float16* w_mprj = (_Float16*)(ws + 1048576);
  _Float16* w_pprj = (_Float16*)(ws + 1179648);
  float*    tab    = (float*)(ws + 1310720);
  float*    scales = (float*)(ws + 1318912);

  const int NQKV = 3 * CDIM * CDIM;  // 196608
  const int NSQ  = CDIM * CDIM;      // 65536
  k_f32_to_f16<<<(NQKV + 255) / 256, 256, 0, stream>>>(m_qkv_w, w_mqkv, NQKV);
  k_f32_to_f16<<<(NQKV + 255) / 256, 256, 0, stream>>>(p_qkv_w, w_pqkv, NQKV);
  k_f32_to_f16<<<(NSQ + 255) / 256, 256, 0, stream>>>(m_anc_w, w_manc, NSQ);
  k_f32_to_f16<<<(NSQ + 255) / 256, 256, 0, stream>>>(p_anc_w, w_panc, NSQ);
  k_f32_to_f16<<<(NSQ + 255) / 256, 256, 0, stream>>>(m_proj_w, w_mprj, NSQ);
  k_f32_to_f16<<<(NSQ + 255) / 256, 256, 0, stream>>>(p_proj_w, w_pprj, NSQ);

  k_cpb<<<(225 * HEADS + 255) / 256, 256, 0, stream>>>(cpb_w1, cpb_b1, cpb_w2, tab);
  k_scales<<<1, 32, 0, stream>>>(m_ls, p_ls, scales);

  (void)hipFuncSetAttribute((const void*)attn_fused,
                            hipFuncAttributeMaxDynamicSharedMemorySize,
                            LDS_BYTES);

  float* out_m = (float*)d_out;
  float* out_p = out_m + (size_t)BWIN * NTOK * CDIM;

  attn_fused<<<BWIN, 256, LDS_BYTES, stream>>>(
      m, m_l, m_s, m_q_bias, m_v_bias, w_mqkv, w_manc, w_mprj, m_proj_b, tab,
      scales, out_m);
  attn_fused<<<BWIN, 256, LDS_BYTES, stream>>>(
      p, p_l, p_s, p_q_bias, p_v_bias, w_pqkv, w_panc, w_pprj, p_proj_b, tab,
      scales + HEADS, out_p);
}
